// QEffDeepseekV3MoE_23871428231438
// MI455X (gfx1250) — matrix operand order repaired and checked by reading, not compiled, not run on hardware
//
#include <hip/hip_runtime.h>
#include <cstdint>
#include <cstddef>

// ---------------- types ----------------
typedef __attribute__((ext_vector_type(16))) __bf16 v16bf;
typedef __attribute__((ext_vector_type(8)))  __bf16 v8bf;
typedef __attribute__((ext_vector_type(2)))  __bf16 v2bf;
typedef __attribute__((ext_vector_type(8)))  float  v8f;

#define LDT 40   // LDS tile row stride in bf16 elems: 80 B -> 16B-aligned b128 frags, bank-spread

// f32x2 -> packed bf16x2 (prefer the hardware pack-convert)
__device__ __forceinline__ unsigned pack2(float a, float b) {
#if __has_builtin(__builtin_amdgcn_cvt_pk_bf16_f32)
  v2bf r = __builtin_amdgcn_cvt_pk_bf16_f32(a, b);
  return __builtin_bit_cast(unsigned, r);
#else
  v2bf r;
  r[0] = (__bf16)a;
  r[1] = (__bf16)b;
  return __builtin_bit_cast(unsigned, r);
#endif
}
// A-fragment K permutation: store K order [0..7,16..23,8..15,24..31] so each lane's
// 16 bf16 (ISA 16-bit A layout, wave32) are one contiguous 32B run in LDS.
__device__ __forceinline__ int permk(int k) {
  return (k & 7) | ((k & 16) >> 1) | ((k & 8) << 1);
}
__device__ __forceinline__ v16bf frag_ld(const unsigned short* p) {
  v8bf lo = *(const v8bf*)(p);
  v8bf hi = *(const v8bf*)(p + 8);
  return __builtin_shufflevector(lo, hi, 0,1,2,3,4,5,6,7,8,9,10,11,12,13,14,15);
}
__device__ __forceinline__ v8f vzero8() {
  v8f z;
#pragma unroll
  for (int i = 0; i < 8; ++i) z[i] = 0.0f;
  return z;
}
__device__ __forceinline__ float silu_f(float g) {
  return g * (1.0f / (1.0f + __expf(-g)));
}

// ---------------- routing tables (deterministic stable bin) ----------------
__global__ void moe_route_kernel(const int* __restrict__ tki, const float* __restrict__ tkw,
                                 int S, int cap,
                                 int* __restrict__ idx_list, int* __restrict__ pos,
                                 float* __restrict__ wsort) {
  const int e = blockIdx.x;           // one wave32 per expert
  const int lane = threadIdx.x;
  int count = 0;
  for (int base = 0; base < S; base += 32) {
    const int s = base + lane;
    const bool mine = (s < S) && (tki[s] == e);
    const unsigned mask = (unsigned)__ballot(mine);
    if (mine) {
      const int r = e * cap + count + __popc(mask & ((1u << lane) - 1u));
      idx_list[r] = s;
      pos[s] = r;
      wsort[r] = tkw[s];
    }
    count += __popc(mask);
  }
}

// ---------------- fused gate+up (+SiLU*up) GEMM, bf16 WMMA core ----------------
// A rows gathered via idx (token = idx[row]/kdiv) or identity when idx==nullptr.
__global__ __launch_bounds__(256)
void moe_gateup_kernel(const float* __restrict__ X, const int* __restrict__ idx,
                       const float* __restrict__ Wg, const float* __restrict__ Wu,
                       float* __restrict__ H,
                       int Mper, int Kdim, int Nn, int kdiv)
{
  __shared__ int sTok[128];
  __shared__ unsigned short sA [2][128 * LDT];
  __shared__ unsigned short sBg[2][128 * LDT];
  __shared__ unsigned short sBu[2][128 * LDT];

  const int tid  = threadIdx.x;
  const int lane = tid & 31;
  const int wv   = tid >> 5;
  const int wm   = wv >> 2;          // 0..1 : 64-row half
  const int wn   = wv & 3;           // 0..3 : 32-col quarter
  const int e    = blockIdx.z;
  const int n0   = blockIdx.x * 128;
  const int rowBase = e * Mper + blockIdx.y * 128;

  const float* __restrict__ wgB = Wg + (size_t)e * Kdim * Nn;
  const float* __restrict__ wuB = Wu + (size_t)e * Kdim * Nn;

  if (tid < 128) {
    const int grow = rowBase + tid;
    sTok[tid] = idx ? (idx[grow] / kdiv) : grow;
  }
  __syncthreads();

  v8f accg[4][2], accu[4][2];
#pragma unroll
  for (int ms = 0; ms < 4; ++ms)
#pragma unroll
    for (int ns = 0; ns < 2; ++ns) { accg[ms][ns] = vzero8(); accu[ms][ns] = vzero8(); }

  const int ar  = tid >> 3;          // A loader: 32 rows/pass, 8 thr x float4 per row
  const int ac  = (tid & 7) << 2;
  const int apc = ac;
  const int bk2 = (tid >> 5) << 1;   // B loader: even k of an adjacent k-pair (0,2,..,14)
  const int bn  = (tid & 31) << 2;   // 4 consecutive n per thread

  float4 aR[4];
  float4 gR[2][2], uR[2][2];         // [k-half pass][row of k-pair]
  const int nk = Kdim >> 5;

  auto g_load = [&](int kb) {
    const int k0 = kb << 5;
#pragma unroll
    for (int p = 0; p < 4; ++p)
      aR[p] = *(const float4*)(X + (size_t)sTok[p * 32 + ar] * Kdim + k0 + ac);
#pragma unroll
    for (int p = 0; p < 2; ++p) {
      const int k = k0 + p * 16 + bk2;
      const size_t o0 = (size_t)k * Nn + n0 + bn;
      const size_t o1 = o0 + Nn;
      gR[p][0] = *(const float4*)(wgB + o0);
      gR[p][1] = *(const float4*)(wgB + o1);
      uR[p][0] = *(const float4*)(wuB + o0);
      uR[p][1] = *(const float4*)(wuB + o1);
    }
  };
  auto l_store = [&](int buf) {
#pragma unroll
    for (int p = 0; p < 4; ++p) {
      unsigned short* pa = &sA[buf][(p * 32 + ar) * LDT + apc];
      *(uint2*)pa = make_uint2(pack2(aR[p].x, aR[p].y), pack2(aR[p].z, aR[p].w));
    }
#pragma unroll
    for (int p = 0; p < 2; ++p) {
      const int k = p * 16 + bk2;    // even -> 4B-aligned b32 stores
      *(unsigned*)&sBg[buf][(bn + 0) * LDT + k] = pack2(gR[p][0].x, gR[p][1].x);
      *(unsigned*)&sBg[buf][(bn + 1) * LDT + k] = pack2(gR[p][0].y, gR[p][1].y);
      *(unsigned*)&sBg[buf][(bn + 2) * LDT + k] = pack2(gR[p][0].z, gR[p][1].z);
      *(unsigned*)&sBg[buf][(bn + 3) * LDT + k] = pack2(gR[p][0].w, gR[p][1].w);
      *(unsigned*)&sBu[buf][(bn + 0) * LDT + k] = pack2(uR[p][0].x, uR[p][1].x);
      *(unsigned*)&sBu[buf][(bn + 1) * LDT + k] = pack2(uR[p][0].y, uR[p][1].y);
      *(unsigned*)&sBu[buf][(bn + 2) * LDT + k] = pack2(uR[p][0].z, uR[p][1].z);
      *(unsigned*)&sBu[buf][(bn + 3) * LDT + k] = pack2(uR[p][0].w, uR[p][1].w);
    }
  };
  auto compute = [&](int buf) {
    const int khalf = (lane >> 4) * 16;
    v16bf af[4];
#pragma unroll
    for (int ms = 0; ms < 4; ++ms)
      af[ms] = frag_ld(&sA[buf][(wm * 64 + ms * 16 + (lane & 15)) * LDT + khalf]);
#pragma unroll
    for (int ns = 0; ns < 2; ++ns) {
      const int c = (wn * 32 + ns * 16 + (lane & 15)) * LDT + khalf;
      v16bf bg = frag_ld(&sBg[buf][c]);
      v16bf bu = frag_ld(&sBu[buf][c]);
#pragma unroll
      for (int ms = 0; ms < 4; ++ms) {
        accg[ms][ns] = __builtin_amdgcn_wmma_f32_16x16x32_bf16(
            false, af[ms], false, bg, (short)0, accg[ms][ns], false, false);
        accu[ms][ns] = __builtin_amdgcn_wmma_f32_16x16x32_bf16(
            false, af[ms], false, bu, (short)0, accu[ms][ns], false, false);
      }
    }
  };

  g_load(0);
  l_store(0);
  __syncthreads();
  for (int kb = 0; kb < nk; ++kb) {
    if (kb + 1 < nk) g_load(kb + 1);
    if (kb + 2 < nk) {
      const size_t poff = (size_t)(((kb + 2) << 5) + bk2) * Nn + n0 + bn;
      __builtin_prefetch(wgB + poff, 0, 1);
      __builtin_prefetch(wuB + poff, 0, 1);
    }
    compute(kb & 1);
    __syncthreads();
    if (kb + 1 < nk) l_store((kb + 1) & 1);
    __syncthreads();
  }

  const int ml = (lane >> 4) * 8;    // C/D layout: lanes16-31 hold M = g+8
  const int nl = lane & 15;
#pragma unroll
  for (int ms = 0; ms < 4; ++ms) {
#pragma unroll
    for (int ns = 0; ns < 2; ++ns) {
      const int gr = rowBase + wm * 64 + ms * 16 + ml;
      const int gc = n0 + wn * 32 + ns * 16 + nl;
#pragma unroll
      for (int g = 0; g < 8; ++g) {
        const float gv = accg[ms][ns][g];
        const float uv = accu[ms][ns][g];
        H[(size_t)(gr + g) * Nn + gc] = silu_f(gv) * uv;
      }
    }
  }
}

// ---------------- down-proj GEMM (optionally row-scaled / accumulate) ----------------
template <bool ACC>
__global__ __launch_bounds__(256)
void moe_down_kernel(const float* __restrict__ Hin, const float* __restrict__ W,
                     const float* __restrict__ scale, float* __restrict__ Out,
                     int Mper, int Kdim, int Nn)
{
  __shared__ float sScale[128];
  __shared__ unsigned short sA[2][128 * LDT];
  __shared__ unsigned short sB[2][128 * LDT];

  const int tid  = threadIdx.x;
  const int lane = tid & 31;
  const int wv   = tid >> 5;
  const int wm   = wv >> 2;
  const int wn   = wv & 3;
  const int e    = blockIdx.z;
  const int n0   = blockIdx.x * 128;
  const int rowBase = e * Mper + blockIdx.y * 128;

  const float* __restrict__ wB = W + (size_t)e * Kdim * Nn;

  if (tid < 128) sScale[tid] = scale ? scale[rowBase + tid] : 1.0f;
  __syncthreads();

  v8f acc[4][2];
#pragma unroll
  for (int ms = 0; ms < 4; ++ms)
#pragma unroll
    for (int ns = 0; ns < 2; ++ns) acc[ms][ns] = vzero8();

  const int ar  = tid >> 3;
  const int ac  = (tid & 7) << 2;
  const int apc = ac;
  const int bk2 = (tid >> 5) << 1;
  const int bn  = (tid & 31) << 2;

  float4 aR[4];
  float4 bR[2][2];
  const int nk = Kdim >> 5;

  auto g_load = [&](int kb) {
    const int k0 = kb << 5;
#pragma unroll
    for (int p = 0; p < 4; ++p)
      aR[p] = *(const float4*)(Hin + (size_t)(rowBase + p * 32 + ar) * Kdim + k0 + ac);
#pragma unroll
    for (int p = 0; p < 2; ++p) {
      const size_t o0 = (size_t)(k0 + p * 16 + bk2) * Nn + n0 + bn;
      bR[p][0] = *(const float4*)(wB + o0);
      bR[p][1] = *(const float4*)(wB + o0 + Nn);
    }
  };
  auto l_store = [&](int buf) {
#pragma unroll
    for (int p = 0; p < 4; ++p) {
      unsigned short* pa = &sA[buf][(p * 32 + ar) * LDT + apc];
      *(uint2*)pa = make_uint2(pack2(aR[p].x, aR[p].y), pack2(aR[p].z, aR[p].w));
    }
#pragma unroll
    for (int p = 0; p < 2; ++p) {
      const int k = p * 16 + bk2;
      *(unsigned*)&sB[buf][(bn + 0) * LDT + k] = pack2(bR[p][0].x, bR[p][1].x);
      *(unsigned*)&sB[buf][(bn + 1) * LDT + k] = pack2(bR[p][0].y, bR[p][1].y);
      *(unsigned*)&sB[buf][(bn + 2) * LDT + k] = pack2(bR[p][0].z, bR[p][1].z);
      *(unsigned*)&sB[buf][(bn + 3) * LDT + k] = pack2(bR[p][0].w, bR[p][1].w);
    }
  };
  auto compute = [&](int buf) {
    const int khalf = (lane >> 4) * 16;
    v16bf af[4];
#pragma unroll
    for (int ms = 0; ms < 4; ++ms)
      af[ms] = frag_ld(&sA[buf][(wm * 64 + ms * 16 + (lane & 15)) * LDT + khalf]);
#pragma unroll
    for (int ns = 0; ns < 2; ++ns) {
      v16bf bf = frag_ld(&sB[buf][(wn * 32 + ns * 16 + (lane & 15)) * LDT + khalf]);
#pragma unroll
      for (int ms = 0; ms < 4; ++ms)
        acc[ms][ns] = __builtin_amdgcn_wmma_f32_16x16x32_bf16(
            false, af[ms], false, bf, (short)0, acc[ms][ns], false, false);
    }
  };

  g_load(0);
  l_store(0);
  __syncthreads();
  for (int kb = 0; kb < nk; ++kb) {
    if (kb + 1 < nk) g_load(kb + 1);
    if (kb + 2 < nk)
      __builtin_prefetch(wB + (size_t)(((kb + 2) << 5) + bk2) * Nn + n0 + bn, 0, 1);
    compute(kb & 1);
    __syncthreads();
    if (kb + 1 < nk) l_store((kb + 1) & 1);
    __syncthreads();
  }

  const int ml = (lane >> 4) * 8;
  const int nl = lane & 15;
#pragma unroll
  for (int ms = 0; ms < 4; ++ms) {
#pragma unroll
    for (int ns = 0; ns < 2; ++ns) {
      const int mr = wm * 64 + ms * 16 + ml;
      const int gc = n0 + wn * 32 + ns * 16 + nl;
#pragma unroll
      for (int g = 0; g < 8; ++g) {
        const float val = acc[ms][ns][g] * sScale[mr + g];
        const size_t o = (size_t)(rowBase + mr + g) * Nn + gc;
        if (ACC) Out[o] += val; else Out[o] = val;
      }
    }
  }
}

// ---------------- deterministic top-k combine ----------------
__global__ void moe_combine_kernel(const float* __restrict__ y, const int* __restrict__ pos,
                                   float* __restrict__ out, int Dn, int K) {
  const int t = blockIdx.x;
  const int d = (int)threadIdx.x << 2;
  float4 a = make_float4(0.f, 0.f, 0.f, 0.f);
  for (int k = 0; k < K; ++k) {
    const int r = pos[t * K + k];
    const float4 v = *(const float4*)(y + (size_t)r * Dn + d);
    a.x += v.x; a.y += v.y; a.z += v.z; a.w += v.w;
  }
  *(float4*)(out + (size_t)t * Dn + d) = a;
}

// ---------------- host launch ----------------
extern "C" void kernel_launch(void* const* d_in, const int* in_sizes, int n_in,
                              void* d_out, int out_size, void* d_ws, size_t ws_size,
                              hipStream_t stream) {
  (void)in_sizes; (void)n_in; (void)out_size; (void)ws_size;
  constexpr int Bc = 2, Lc = 2048, Dc = 2048, Ec = 32, Kc = 4, Fc = 1408, FSc = 2 * 1408;
  constexpr int Tc = Bc * Lc, Sc = Tc * Kc, capc = Sc / Ec;

  const float* X   = (const float*)d_in[0];
  const int*   tki = (const int*)  d_in[1];
  const float* tkw = (const float*)d_in[2];
  const float* Wg  = (const float*)d_in[3];
  const float* Wu  = (const float*)d_in[4];
  const float* Wd  = (const float*)d_in[5];
  const float* Wsg = (const float*)d_in[6];
  const float* Wsu = (const float*)d_in[7];
  const float* Wsd = (const float*)d_in[8];
  float* out = (float*)d_out;

  char* ws = (char*)d_ws;
  int*   idx_list = (int*)  (ws + 0);
  int*   pos      = (int*)  (ws + 65536);
  float* wsort    = (float*)(ws + 131072);
  float* h        = (float*)(ws + 196608);                       // [S, F]
  float* y        = (float*)(ws + 92471296ULL);                  // [S, D]
  float* hs       = (float*)(ws + 226689024ULL);                 // [T, FS]

  // 1) routing tables (stable, deterministic)
  moe_route_kernel<<<Ec, 32, 0, stream>>>(tki, tkw, Sc, capc, idx_list, pos, wsort);

  // 2) routed experts: h = silu(Xe*Wg) * (Xe*Wu)
  moe_gateup_kernel<<<dim3(Fc / 128, capc / 128, Ec), 256, 0, stream>>>(
      X, idx_list, Wg, Wu, h, capc, Dc, Fc, Kc);

  // 3) routed experts: y = (h * Wd) scaled by sorted top-k weight
  moe_down_kernel<false><<<dim3(Dc / 128, capc / 128, Ec), 256, 0, stream>>>(
      h, Wd, wsort, y, capc, Fc, Dc);

  // 4) shared expert gate/up
  moe_gateup_kernel<<<dim3(FSc / 128, Tc / 128, 1), 256, 0, stream>>>(
      X, nullptr, Wsg, Wsu, hs, Tc, Dc, FSc, 1);

  // 5) combine routed contributions into out (write)
  moe_combine_kernel<<<Tc, 512, 0, stream>>>(y, pos, out, Dc, Kc);

  // 6) shared expert down, accumulated into out
  moe_down_kernel<true><<<dim3(Dc / 128, Tc / 128, 1), 256, 0, stream>>>(
      hs, Wsd, nullptr, out, Tc, FSc, Dc);
}
